// CausalSelfAttention_45397804319448
// MI455X (gfx1250) — matrix operand order repaired and checked by reading, not compiled, not run on hardware
//
#include <hip/hip_runtime.h>
#include <hip/hip_bf16.h>

typedef __bf16 v16bf __attribute__((ext_vector_type(16)));
typedef __bf16 v8bf  __attribute__((ext_vector_type(8)));
typedef float  v8f   __attribute__((ext_vector_type(8)));

constexpr int Bc = 2, Tc = 4096, Dc = 768, Hc = 12, DKc = 64;
constexpr int Mrows = Bc * Tc;      // 8192
constexpr int N3    = 3 * Dc;       // 2304
constexpr int Tpad  = Tc + 64;      // padded K/V rows so tail tiles never fault

// ---------------------------------------------------------------------------
// Fragment loaders (CDNA5 WMMA 16x16x32 bf16 layouts, CDNA5 ISA)
// A 16x32 (MxK): lane L -> M = L&15, half = L>>4; elems 0..7 = K[8*half .. +7],
//                elems 8..15 = K[16+8*half .. +7]  -> two contiguous 16B loads.
// B 32x16 (KxN): lane L -> N = L&15, half = L>>4; elems 0..7 = K[8*half .. +7],
//                elems 8..15 = K[16+8*half .. +7]  -> two 16B loads when K-major.
// C/D 16x16 f32: lane L -> N = L&15; VGPR r -> M = r + 8*(L>>4).
// ---------------------------------------------------------------------------
__device__ inline v16bf load_a_frag(const __bf16* base, int lane, int ld) {
  int m = lane & 15, hf = lane >> 4;
  const __bf16* p = base + (size_t)m * ld + 8 * hf;
  v8bf lo = *(const v8bf*)p;
  v8bf hi = *(const v8bf*)(p + 16);
  return __builtin_shufflevector(lo, hi, 0,1,2,3,4,5,6,7,8,9,10,11,12,13,14,15);
}

__device__ inline v16bf load_b_frag(const __bf16* base, int lane, int ld) {
  int n = lane & 15, hf = lane >> 4;
  const __bf16* p = base + (size_t)n * ld + 8 * hf;
  v8bf lo = *(const v8bf*)p;
  v8bf hi = *(const v8bf*)(p + 16);
  return __builtin_shufflevector(lo, hi, 0,1,2,3,4,5,6,7,8,9,10,11,12,13,14,15);
}

__device__ inline v8f wmma_bf16(v16bf a, v16bf b, v8f c) {
  return __builtin_amdgcn_wmma_f32_16x16x32_bf16(
      /*neg_a=*/false, a, /*neg_b=*/false, b,
      /*c_mod=*/(short)0, c, /*reuse_a=*/false, /*reuse_b=*/false);
}

// ---------------------------------------------------------------------------
// Prep: fp32 -> bf16 convert / transpose / pad-zero
// ---------------------------------------------------------------------------
__global__ void convert_x_kernel(const float* __restrict__ x,
                                 __bf16* __restrict__ xb, int n) {
  int i = blockIdx.x * 256 + threadIdx.x;
  if (i < n) xb[i] = (__bf16)x[i];
}

// W is [K][N] row-major fp32; Wt is [N][K] bf16 (K contiguous for B-fragments)
__global__ void transpose_w_kernel(const float* __restrict__ W,
                                   __bf16* __restrict__ Wt, int K, int N) {
  int i = blockIdx.x * 256 + threadIdx.x;
  if (i < K * N) {
    int n = i / K, k = i - n * K;
    Wt[i] = (__bf16)W[(size_t)k * N + n];
  }
}

__global__ void zero_pad_kernel(__bf16* __restrict__ kb, __bf16* __restrict__ vT) {
  int i = blockIdx.x * 256 + threadIdx.x;           // Bc*Hc * 64 * DKc threads
  int bh = i >> 12, r = i & 4095, trel = r >> 6, d = r & 63;
  kb[((size_t)bh * Tpad + Tc + trel) * DKc + d] = (__bf16)0.f;
  vT[((size_t)bh * DKc + d) * Tpad + Tc + trel]  = (__bf16)0.f;
}

// ---------------------------------------------------------------------------
// QKV GEMM: [8192 x 768] x [768 x 2304] + bias.
// Wave computes a 64x64 tile.  Q,K scattered to [B,H,T(,pad),DK];
// V stored transposed to [B,H,DK,Tpad] with one packed b128 store per tile.
// ---------------------------------------------------------------------------
__global__ __launch_bounds__(128) void qkv_gemm_kernel(
    const __bf16* __restrict__ xb, const __bf16* __restrict__ wt,
    const float* __restrict__ bias,
    __bf16* __restrict__ qb, __bf16* __restrict__ kb, __bf16* __restrict__ vT) {
  int lane = threadIdx.x & 31, wv = threadIdx.x >> 5;
  int m0 = blockIdx.x * 64;
  int n0 = blockIdx.y * 256 + wv * 64;

  v8f acc[4][4] = {};
  for (int k = 0; k < Dc; k += 32) {
    v16bf a[4], bb[4];
#pragma unroll
    for (int mt = 0; mt < 4; mt++)
      a[mt] = load_a_frag(xb + (size_t)(m0 + mt * 16) * Dc + k, lane, Dc);
#pragma unroll
    for (int nt = 0; nt < 4; nt++)
      bb[nt] = load_b_frag(wt + (size_t)(n0 + nt * 16) * Dc + k, lane, Dc);
#pragma unroll
    for (int mt = 0; mt < 4; mt++)
#pragma unroll
      for (int nt = 0; nt < 4; nt++)
        acc[mt][nt] = wmma_bf16(a[mt], bb[nt], acc[mt][nt]);
  }

  int nlane = lane & 15, hf = lane >> 4;
#pragma unroll
  for (int mt = 0; mt < 4; mt++) {
    int mg0 = m0 + mt * 16 + 8 * hf;
    int bidx = mg0 >> 12, t0 = mg0 & (Tc - 1);
#pragma unroll
    for (int nt = 0; nt < 4; nt++) {
      int nglob = n0 + nt * 16 + nlane;
      float bv = bias[nglob];
      int sec = nglob / Dc, col = nglob % Dc;
      int h = col >> 6, d = col & 63;
      v8f acv = acc[mt][nt];
      if (sec == 2) {                       // V: transposed packed store
        v8bf pk;
#pragma unroll
        for (int r = 0; r < 8; r++) pk[r] = (__bf16)(acv[r] + bv);
        *(v8bf*)(vT + ((size_t)(bidx * Hc + h) * DKc + d) * Tpad + t0) = pk;
      } else {                              // Q / K: strided scalar stores
        __bf16* dst = (sec == 0)
            ? qb + ((size_t)(bidx * Hc + h) * Tc + t0) * DKc + d
            : kb + ((size_t)(bidx * Hc + h) * Tpad + t0) * DKc + d;
#pragma unroll
        for (int r = 0; r < 8; r++) dst[(size_t)r * DKc] = (__bf16)(acv[r] + bv);
      }
    }
  }
}

// ---------------------------------------------------------------------------
// Flash attention: wave = 16 query rows, online softmax over 32-col steps.
// ---------------------------------------------------------------------------
__global__ __launch_bounds__(128) void attn_kernel(
    const __bf16* __restrict__ qb, const __bf16* __restrict__ kb,
    const __bf16* __restrict__ vT, __bf16* __restrict__ attn) {
  __shared__ __align__(16) __bf16 pbuf[4][16 * 32];   // per-wave P staging
  int lane = threadIdx.x & 31, wv = threadIdx.x >> 5;
  int bh = blockIdx.y;
  int b = bh / Hc, h = bh - b * Hc;
  int t0 = blockIdx.x * 64 + wv * 16;                 // first query row (in-batch)

  const __bf16* qrow = qb + (size_t)bh * Tc * DKc;
  const __bf16* krow = kb + (size_t)bh * Tpad * DKc;
  const __bf16* vrow = vT + (size_t)bh * DKc * Tpad;

  v16bf qa0 = load_a_frag(qrow + (size_t)t0 * DKc + 0,  lane, DKc);
  v16bf qa1 = load_a_frag(qrow + (size_t)t0 * DKc + 32, lane, DKc);

  v8f o[4] = {};
  float mrow[8], lrow[8];
#pragma unroll
  for (int r = 0; r < 8; r++) { mrow[r] = -1e30f; lrow[r] = 0.f; }

  const float scale = 0.125f;               // 1/sqrt(64)
  int nlane = lane & 15, hf = lane >> 4;
  int qi0 = t0 + 8 * hf;

  for (int j0 = 0; j0 < t0 + 16; j0 += 32) {
    v8f s0 = {}, s1 = {};
    s0 = wmma_bf16(qa0, load_b_frag(krow + (size_t)j0 * DKc + 0,  lane, DKc), s0);
    s0 = wmma_bf16(qa1, load_b_frag(krow + (size_t)j0 * DKc + 32, lane, DKc), s0);
    s1 = wmma_bf16(qa0, load_b_frag(krow + (size_t)(j0 + 16) * DKc + 0,  lane, DKc), s1);
    s1 = wmma_bf16(qa1, load_b_frag(krow + (size_t)(j0 + 16) * DKc + 32, lane, DKc), s1);

    float p0[8], p1[8], corr[8];
#pragma unroll
    for (int r = 0; r < 8; r++) {
      float v0 = s0[r] * scale, v1 = s1[r] * scale;
      int qi = qi0 + r;
      v0 = (j0 + nlane      > qi) ? -1e30f : v0;      // causal mask
      v1 = (j0 + 16 + nlane > qi) ? -1e30f : v1;
      float rm = fmaxf(v0, v1);
#pragma unroll
      for (int sh = 1; sh < 16; sh <<= 1) rm = fmaxf(rm, __shfl_xor(rm, sh, 32));
      float mnew = fmaxf(mrow[r], rm);
      corr[r] = __expf(mrow[r] - mnew);
      p0[r] = __expf(v0 - mnew);
      p1[r] = __expf(v1 - mnew);
      float rs = p0[r] + p1[r];
#pragma unroll
      for (int sh = 1; sh < 16; sh <<= 1) rs += __shfl_xor(rs, sh, 32);
      mrow[r] = mnew;
      lrow[r] = lrow[r] * corr[r] + rs;
    }
#pragma unroll
    for (int dt = 0; dt < 4; dt++)
#pragma unroll
      for (int r = 0; r < 8; r++) o[dt][r] *= corr[r];

    // stage P (16x32, row-major bf16) through LDS: C layout -> A layout
#pragma unroll
    for (int r = 0; r < 8; r++) {
      pbuf[wv][(r + 8 * hf) * 32 + nlane]      = (__bf16)p0[r];
      pbuf[wv][(r + 8 * hf) * 32 + 16 + nlane] = (__bf16)p1[r];
    }
    v16bf pa = load_a_frag(&pbuf[wv][0], lane, 32);
#pragma unroll
    for (int dt = 0; dt < 4; dt++) {
      v16bf vb = load_b_frag(vrow + (size_t)(dt * 16) * Tpad + j0, lane, Tpad);
      o[dt] = wmma_bf16(pa, vb, o[dt]);
    }
  }

#pragma unroll
  for (int dt = 0; dt < 4; dt++)
#pragma unroll
    for (int r = 0; r < 8; r++) {
      float val = o[dt][r] / lrow[r];
      size_t mg = (size_t)(b * Tc + t0 + r + 8 * hf);
      attn[mg * Dc + h * DKc + dt * 16 + nlane] = (__bf16)val;
    }
}

// ---------------------------------------------------------------------------
// Output projection: [8192 x 768] x [768 x 768] + bias -> fp32 d_out
// ---------------------------------------------------------------------------
__global__ __launch_bounds__(128) void proj_gemm_kernel(
    const __bf16* __restrict__ ab, const __bf16* __restrict__ wt,
    const float* __restrict__ bias, float* __restrict__ out) {
  int lane = threadIdx.x & 31, wv = threadIdx.x >> 5;
  int m0 = blockIdx.x * 64;
  int n0 = blockIdx.y * 256 + wv * 64;

  v8f acc[4][4] = {};
  for (int k = 0; k < Dc; k += 32) {
    v16bf a[4], bb[4];
#pragma unroll
    for (int mt = 0; mt < 4; mt++)
      a[mt] = load_a_frag(ab + (size_t)(m0 + mt * 16) * Dc + k, lane, Dc);
#pragma unroll
    for (int nt = 0; nt < 4; nt++)
      bb[nt] = load_b_frag(wt + (size_t)(n0 + nt * 16) * Dc + k, lane, Dc);
#pragma unroll
    for (int mt = 0; mt < 4; mt++)
#pragma unroll
      for (int nt = 0; nt < 4; nt++)
        acc[mt][nt] = wmma_bf16(a[mt], bb[nt], acc[mt][nt]);
  }

  int nlane = lane & 15, hf = lane >> 4;
#pragma unroll
  for (int mt = 0; mt < 4; mt++) {
    int mg0 = m0 + mt * 16 + 8 * hf;
#pragma unroll
    for (int nt = 0; nt < 4; nt++) {
      int nglob = n0 + nt * 16 + nlane;
      float bv = bias[nglob];
#pragma unroll
      for (int r = 0; r < 8; r++)
        out[(size_t)(mg0 + r) * Dc + nglob] = acc[mt][nt][r] + bv;
    }
  }
}

// ---------------------------------------------------------------------------
extern "C" void kernel_launch(void* const* d_in, const int* in_sizes, int n_in,
                              void* d_out, int out_size, void* d_ws, size_t ws_size,
                              hipStream_t stream) {
  const float* x     = (const float*)d_in[0];
  const float* Wqkv  = (const float*)d_in[1];
  const float* bqkv  = (const float*)d_in[2];
  const float* Wproj = (const float*)d_in[3];
  const float* bproj = (const float*)d_in[4];
  float* out = (float*)d_out;

  char* ws = (char*)d_ws;
  size_t off = 0;
  auto carve = [&](size_t bytes) -> char* {
    char* p = ws + off;
    off += (bytes + 255) & ~(size_t)255;
    return p;
  };
  __bf16* xb   = (__bf16*)carve((size_t)Mrows * Dc * 2);
  __bf16* wqt  = (__bf16*)carve((size_t)N3 * Dc * 2);
  __bf16* wpt  = (__bf16*)carve((size_t)Dc * Dc * 2);
  __bf16* qbuf = (__bf16*)carve((size_t)Bc * Hc * Tc * DKc * 2);
  __bf16* kbuf = (__bf16*)carve((size_t)Bc * Hc * Tpad * DKc * 2);
  __bf16* vTb  = (__bf16*)carve((size_t)Bc * Hc * DKc * Tpad * 2);
  __bf16* attn = (__bf16*)carve((size_t)Mrows * Dc * 2);

  convert_x_kernel<<<(Mrows * Dc + 255) / 256, 256, 0, stream>>>(x, xb, Mrows * Dc);
  transpose_w_kernel<<<(Dc * N3 + 255) / 256, 256, 0, stream>>>(Wqkv, wqt, Dc, N3);
  transpose_w_kernel<<<(Dc * Dc + 255) / 256, 256, 0, stream>>>(Wproj, wpt, Dc, Dc);
  zero_pad_kernel<<<(Bc * Hc * 64 * DKc + 255) / 256, 256, 0, stream>>>(kbuf, vTb);

  qkv_gemm_kernel<<<dim3(Mrows / 64, N3 / 256), 128, 0, stream>>>(
      xb, wqt, bqkv, qbuf, kbuf, vTb);
  attn_kernel<<<dim3(Tc / 64, Bc * Hc), 128, 0, stream>>>(qbuf, kbuf, vTb, attn);
  proj_gemm_kernel<<<dim3(Mrows / 64, Dc / 256), 128, 0, stream>>>(
      attn, wpt, bproj, out);
}
